// Attention_83167746720282
// MI455X (gfx1250) — matrix operand order repaired and checked by reading, not compiled, not run on hardware
//
#include <hip/hip_runtime.h>
#include <hip/hip_bf16.h>
#include <math.h>

// Problem constants
#define B_    32
#define C_    684
#define H_    16
#define W_    64
#define HW_   1024
#define HID_  256
#define A_    512
#define CK_   512
#define KK_   11
#define TAPS_ 121

// GEMM K padding: coverage K 121 -> 128 (4 k-steps of 32), cnn K 684 -> 704 (22 k-steps)
#define KS_COV 4
#define KS_CNN 22

typedef __attribute__((ext_vector_type(16))) __bf16          v16bf;
typedef __attribute__((ext_vector_type(8)))  float           v8f;
typedef __attribute__((ext_vector_type(8)))  unsigned short  u16x8;
typedef __attribute__((ext_vector_type(16))) unsigned short  u16x16;
typedef int v4i_ __attribute__((vector_size(16)));

// Workspace layout (bytes); total ~1.05 MB
#define QB_OFF    0u        // query+bh+bec folded: 32*512 f32 = 64 KB
#define EN_OFF    65536u    // energy: 32*1024 f32 = 128 KB
#define GMAX_OFF  196608u   // global max scalar (256 B slot)
#define WCOMB_OFF 196864u   // folded coverage weight, swizzled bf16: 4*32*32*16 = 128 KB
#define WEC_OFF   327936u   // Wec^T swizzled bf16: 22*32*32*16 = 704 KB

__device__ __forceinline__ unsigned short f2bf(float x) {
  unsigned u = __builtin_bit_cast(unsigned, x);
  u += 0x7fffu + ((u >> 16) & 1u);   // round-to-nearest-even
  return (unsigned short)(u >> 16);
}

// Async global->LDS 16B copy (CDNA5 GLOBAL_LOAD_ASYNC_TO_LDS_B128, ASYNCcnt).
__device__ __forceinline__ void async_load_b128(const float* g, float* lds_generic) {
#if __has_builtin(__builtin_amdgcn_global_load_async_to_lds_b128)
  __builtin_amdgcn_global_load_async_to_lds_b128(
      (__attribute__((address_space(1))) v4i_*)(uintptr_t)g,
      (__attribute__((address_space(3))) v4i_*)(uintptr_t)lds_generic,
      0, 0);
#else
  unsigned int loff = (unsigned int)(uintptr_t)lds_generic;   // LDS byte address
  unsigned long long ga = (unsigned long long)(uintptr_t)g;
  asm volatile("global_load_async_to_lds_b128 %0, %1, off"
               :: "v"(loff), "v"(ga) : "memory");
#endif
}

__device__ __forceinline__ void wait_async0() {
#if __has_builtin(__builtin_amdgcn_s_wait_asynccnt)
  __builtin_amdgcn_s_wait_asynccnt(0);
#else
  asm volatile("s_wait_asynccnt 0x0" ::: "memory");
#endif
}

// ---------------------------------------------------------------------------
// qb[b,a] = hidden[b,:] . Wh[a,:] + bh[a] + bec[a]
__global__ __launch_bounds__(256) void prep_query(
    const float* __restrict__ hidden, const float* __restrict__ Wh,
    const float* __restrict__ bh, const float* __restrict__ bec,
    float* __restrict__ qb) {
  int idx = blockIdx.x * 256 + threadIdx.x;      // B_*A_ = 16384
  int b = idx >> 9, a = idx & (A_ - 1);
  const float* hv = hidden + b * HID_;
  const float* wr = Wh + a * HID_;
  float s = 0.f;
#pragma unroll 4
  for (int h = 0; h < HID_; ++h) s += hv[h] * wr[h];
  qb[idx] = s + bh[a] + bec[a];
}

// ---------------------------------------------------------------------------
// Wcomb[tap,a] = sum_k Waw[a,k]*Wac[k,tap], stored as bf16 in WMMA B-fragment
// swizzle: frag f = ks*32+nt, per lane 16 K values (8h+s, then 16+8h+s-8) for column n.
__global__ __launch_bounds__(256) void prep_wcomb(
    const float* __restrict__ Waw, const float* __restrict__ Wac,
    unsigned short* __restrict__ dst) {
  int idx  = blockIdx.x * 256 + threadIdx.x;     // 4*32 frags * 512 = 65536
  int i    = idx & 15;
  int lane = (idx >> 4) & 31;
  int f    = idx >> 9;                            // ks*32 + nt
  int ks = f >> 5, nt = f & 31;
  int a = nt * 16 + (lane & 15);
  int k = ks * 32 + ((i >> 3) << 4) + ((lane >> 4) << 3) + (i & 7);  // B layout: K 8h+s / 16+8h+s-8 per half-wave
  float s = 0.f;
  if (k < TAPS_) {
    const float* wa = Waw + a * CK_;
#pragma unroll 4
    for (int ck = 0; ck < CK_; ++ck) s += wa[ck] * Wac[ck * TAPS_ + k];
  }
  dst[idx] = f2bf(s);
}

// ---------------------------------------------------------------------------
// Wec^T (c-major) in bf16 B-fragment swizzle, K padded 684->704.
__global__ __launch_bounds__(256) void prep_wec(
    const float* __restrict__ Wec, unsigned short* __restrict__ dst) {
  int idx  = blockIdx.x * 256 + threadIdx.x;     // 22*32 frags * 512 = 360448
  int i    = idx & 15;
  int lane = (idx >> 4) & 31;
  int f    = idx >> 9;                            // ks*32 + nt
  int ks = f >> 5, nt = f & 31;
  int a = nt * 16 + (lane & 15);
  int c = ks * 32 + ((i >> 3) << 4) + ((lane >> 4) << 3) + (i & 7);
  float v = (c < C_) ? Wec[a * C_ + c] : 0.f;
  dst[idx] = f2bf(v);
}

// ---------------------------------------------------------------------------
// Fused: score = tanh(qb + patch@Wcomb + feat@Wec^T); energy = score . Wv + bv
// Grid: (HW/16, B). Block: 256 threads = 8 wave32s; wave handles 4 N-tiles.
// cnn_features slab is fetched with async global->LDS copies overlapped with
// the coverage GEMM, then converted to bf16 A-fragment layout.
#define FSTRIDE 736   // 704 + pad, bf16 elements, 16B aligned rows
#define PSTRIDE 160   // 128 + pad
__global__ __launch_bounds__(256) void energy_kernel(
    const float* __restrict__ cnn, const float* __restrict__ asum,
    const float* __restrict__ Wv, const float* __restrict__ bv,
    const float* __restrict__ qb,
    const unsigned short* __restrict__ wcomb,
    const unsigned short* __restrict__ wec,
    float* __restrict__ energy) {
  __shared__ float          sSlab[704 * 16];        // f32 staging, [c][t], 45056 B
  __shared__ unsigned short sFeat[16 * FSTRIDE];    // bf16 A-operand
  __shared__ unsigned short sPatch[16 * PSTRIDE];
  __shared__ float sE[16];

  const int tid  = threadIdx.x;
  const int b    = blockIdx.y;
  const int hw0  = blockIdx.x << 4;
  const int lane = tid & 31;
  const int wv   = tid >> 5;

  if (tid < 16) sE[tid] = 0.f;

  // Kick off async fetch of the 16x684 f32 feature slab (ASYNCcnt),
  // 16B per transfer: slab[c*16 + q*4 ..] = cnn[b, c, hw0 + q*4 ..].
  for (int idx = tid; idx < C_ * 4; idx += 256) {
    int c = idx >> 2, q = idx & 3;
    async_load_b128(cnn + ((size_t)(b * C_ + c) << 10) + hw0 + (q << 2),
                    &sSlab[(c << 4) + (q << 2)]);
  }

  // Zero-pad bf16 slab K region [684, 704).
  for (int idx = tid; idx < 16 * (704 - C_); idx += 256) {
    int t = idx & 15, c = C_ + (idx >> 4);
    sFeat[t * FSTRIDE + c] = 0;
  }
  // Stage 11x11 zero-padded alpha_sum patches (K=121 -> 128), as bf16.
  for (int idx = tid; idx < 16 * 128; idx += 256) {
    int t = idx >> 7, j = idx & 127;
    unsigned short v = 0;
    if (j < TAPS_) {
      int hw = hw0 + t;
      int h = hw >> 6, w = hw & 63;
      int hh = h + j / KK_ - 5, ww = w + j % KK_ - 5;
      if (hh >= 0 && hh < H_ && ww >= 0 && ww < W_)
        v = f2bf(asum[(b << 10) + (hh << 6) + ww]);
    }
    sPatch[t * PSTRIDE + j] = v;
  }
  __syncthreads();

  const int nLow   = lane & 15;
  const int rowSel = lane >> 4;       // half-wave select
  const int koffA  = rowSel << 3;     // A-frag K split: 0..7 / 8..15 per half

  // Init accumulators with qb (query + bh + bec), constant per column.
  v8f acc[4];
#pragma unroll
  for (int j = 0; j < 4; ++j) {
    float q = qb[b * A_ + ((wv * 4 + j) << 4) + nLow];
    acc[j] = (v8f){q, q, q, q, q, q, q, q};
  }

  // Coverage GEMM: M16 x K128 x N512 — overlaps with the async feature fetch.
#pragma unroll
  for (int ks = 0; ks < KS_COV; ++ks) {
    const unsigned short* row = sPatch + nLow * PSTRIDE + ks * 32;
    u16x8 c0 = *(const u16x8*)(row + koffA);
    u16x8 c1 = *(const u16x8*)(row + 16 + koffA);
    u16x16 at;
#pragma unroll
    for (int i = 0; i < 8; ++i) { at[i] = c0[i]; at[i + 8] = c1[i]; }
    v16bf afrag = __builtin_bit_cast(v16bf, at);
#pragma unroll
    for (int j = 0; j < 4; ++j) {
      int nt = wv * 4 + j;
      u16x16 bt = *(const u16x16*)(wcomb + (((ks * 32 + nt) * 32 + lane) << 4));
      v16bf bfrag = __builtin_bit_cast(v16bf, bt);
      acc[j] = __builtin_amdgcn_wmma_f32_16x16x32_bf16(
          false, afrag, false, bfrag, (short)0, acc[j], false, false);
    }
  }

  // Drain async copies, then transpose-convert f32 slab -> bf16 A layout.
  wait_async0();
  __syncthreads();
  for (int idx = tid; idx < 16 * C_; idx += 256) {
    int t = idx & 15, c = idx >> 4;
    sFeat[t * FSTRIDE + c] = f2bf(sSlab[(c << 4) + t]);
  }
  __syncthreads();

  // cnn_trans GEMM: M16 x K704 x N512
  for (int ks = 0; ks < KS_CNN; ++ks) {
    const unsigned short* row = sFeat + nLow * FSTRIDE + ks * 32;
    u16x8 c0 = *(const u16x8*)(row + koffA);
    u16x8 c1 = *(const u16x8*)(row + 16 + koffA);
    u16x16 at;
#pragma unroll
    for (int i = 0; i < 8; ++i) { at[i] = c0[i]; at[i + 8] = c1[i]; }
    v16bf afrag = __builtin_bit_cast(v16bf, at);
    if (ks + 1 < KS_CNN)
      __builtin_prefetch(wec + ((((ks + 1) * 32 + wv * 4) * 32 + lane) << 4), 0, 0);
#pragma unroll
    for (int j = 0; j < 4; ++j) {
      int nt = wv * 4 + j;
      u16x16 bt = *(const u16x16*)(wec + (((ks * 32 + nt) * 32 + lane) << 4));
      v16bf bfrag = __builtin_bit_cast(v16bf, bt);
      acc[j] = __builtin_amdgcn_wmma_f32_16x16x32_bf16(
          false, afrag, false, bfrag, (short)0, acc[j], false, false);
    }
  }

  // Epilogue: energy[m] += tanh(score[m,n]) * Wv[n]; cross-lane reduce then LDS atomics.
  float s[8] = {0, 0, 0, 0, 0, 0, 0, 0};
#pragma unroll
  for (int j = 0; j < 4; ++j) {
    float wvx = Wv[((wv * 4 + j) << 4) + nLow];
#pragma unroll
    for (int r = 0; r < 8; ++r) s[r] += tanhf(acc[j][r]) * wvx;
  }
#pragma unroll
  for (int r = 0; r < 8; ++r) {
    float v = s[r];
    v += __shfl_xor(v, 1);
    v += __shfl_xor(v, 2);
    v += __shfl_xor(v, 4);
    v += __shfl_xor(v, 8);
    if (nLow == 0) atomicAdd(&sE[(rowSel << 3) + r], v);
  }
  __syncthreads();
  if (tid < 16) energy[(b << 10) + hw0 + tid] = sE[tid] + bv[0];
}

// ---------------------------------------------------------------------------
__global__ __launch_bounds__(1024) void maxred_kernel(
    const float* __restrict__ energy, float* __restrict__ gmax) {
  __shared__ float red[1024];
  int tid = threadIdx.x;
  float v = -3.402823466e38f;
  for (int i = tid; i < B_ * HW_; i += 1024) v = fmaxf(v, energy[i]);
  red[tid] = v;
  __syncthreads();
  for (int off = 512; off > 0; off >>= 1) {
    if (tid < off) red[tid] = fmaxf(red[tid], red[tid + off]);
    __syncthreads();
  }
  if (tid == 0) *gmax = red[0];
}

// ---------------------------------------------------------------------------
// Per-batch masked softmax; writes alpha and alpha_sum_new outputs.
__global__ __launch_bounds__(1024) void softmax_kernel(
    const float* __restrict__ energy, const float* __restrict__ mask,
    const float* __restrict__ asum, const float* __restrict__ gmax,
    float* __restrict__ outAlpha, float* __restrict__ outAsum) {
  __shared__ float red[1024];
  int b = blockIdx.x, t = threadIdx.x;
  float e = expf(energy[(b << 10) + t] - gmax[0]) * mask[(b << 10) + t];
  red[t] = e;
  __syncthreads();
  for (int off = 512; off > 0; off >>= 1) {
    if (t < off) red[t] += red[t + off];
    __syncthreads();
  }
  float alpha = e / (red[0] + 1e-10f);
  outAlpha[(b << 10) + t] = alpha;
  outAsum[(b << 10) + t] = alpha + asum[(b << 10) + t];
}

// ---------------------------------------------------------------------------
// context[b,c] = sum_hw alpha[b,hw] * cnn[b,c,hw]; 64 threads per c, coalesced.
__global__ __launch_bounds__(256) void context_kernel(
    const float* __restrict__ alpha, const float* __restrict__ cnn,
    float* __restrict__ ctx) {
  __shared__ float red[256];
  int b = blockIdx.x;
  int c = blockIdx.y * 4 + (threadIdx.x >> 6);
  int l = threadIdx.x & 63;
  const float* ap = alpha + (b << 10);
  const float* fp = cnn + ((size_t)(b * C_ + c) << 10);
  float p = 0.f;
  for (int hw = l; hw < HW_; hw += 64) p += ap[hw] * fp[hw];
  red[threadIdx.x] = p;
  __syncthreads();
  for (int off = 32; off > 0; off >>= 1) {
    if (l < off) red[threadIdx.x] += red[threadIdx.x + off];
    __syncthreads();
  }
  if (l == 0 && c < C_) ctx[b * C_ + c] = red[threadIdx.x];
}

// ---------------------------------------------------------------------------
extern "C" void kernel_launch(void* const* d_in, const int* in_sizes, int n_in,
                              void* d_out, int out_size, void* d_ws, size_t ws_size,
                              hipStream_t stream) {
  const float* cnn    = (const float*)d_in[0];
  const float* hidden = (const float*)d_in[1];
  const float* asum   = (const float*)d_in[2];
  const float* mask   = (const float*)d_in[3];
  const float* Wh     = (const float*)d_in[4];
  const float* bh     = (const float*)d_in[5];
  const float* Wec    = (const float*)d_in[6];
  const float* bec    = (const float*)d_in[7];
  const float* Wac    = (const float*)d_in[8];
  const float* Waw    = (const float*)d_in[9];
  const float* Wv     = (const float*)d_in[10];
  const float* bv     = (const float*)d_in[11];

  char* ws = (char*)d_ws;
  float*          qb     = (float*)(ws + QB_OFF);
  float*          energy = (float*)(ws + EN_OFF);
  float*          gmax   = (float*)(ws + GMAX_OFF);
  unsigned short* wcomb  = (unsigned short*)(ws + WCOMB_OFF);
  unsigned short* wecs   = (unsigned short*)(ws + WEC_OFF);

  float* outCtx   = (float*)d_out;             // [B, C]
  float* outAlpha = outCtx + B_ * C_;          // [B, H, W]
  float* outAsum  = outAlpha + B_ * HW_;       // [B, 1, H, W]

  prep_query<<<64, 256, 0, stream>>>(hidden, Wh, bh, bec, qb);
  prep_wcomb<<<256, 256, 0, stream>>>(Waw, Wac, wcomb);
  prep_wec<<<1408, 256, 0, stream>>>(Wec, wecs);
  energy_kernel<<<dim3(HW_ / 16, B_), 256, 0, stream>>>(
      cnn, asum, Wv, bv, qb, wcomb, wecs, energy);
  maxred_kernel<<<1, 1024, 0, stream>>>(energy, gmax);
  softmax_kernel<<<B_, 1024, 0, stream>>>(energy, mask, asum, gmax, outAlpha, outAsum);
  context_kernel<<<dim3(B_, C_ / 4), 256, 0, stream>>>(outAlpha, cnn, outCtx);
}
